// HiNT_24524263260435
// MI455X (gfx1250) — matrix operand order repaired and checked by reading, not compiled, not run on hardware
//
#include <hip/hip_runtime.h>

#define VOCAB  50000
#define EMB    300
#define BB     64
#define MAXDOC 800
#define QQ     10
#define PP     8
#define LSTMH  6
#define KKMAX  10

typedef __attribute__((ext_vector_type(16))) _Float16 v16h;
typedef __attribute__((ext_vector_type(8)))  float    v8f;

__device__ __forceinline__ float shflx(float v, int xmask) {
    int l = (((int)(threadIdx.x & 31)) ^ xmask) << 2;
    int r = __builtin_amdgcn_ds_bpermute(l, __float_as_int(v));
    return __int_as_float(r);
}
// fast reciprocal: single v_rcp_f32 (TRANS pipe, co-executes with VALU)
__device__ __forceinline__ float frcp(float x)     { return __builtin_amdgcn_rcpf(x); }
__device__ __forceinline__ float sigf(float x)     { return frcp(1.0f + __expf(-x)); }
__device__ __forceinline__ float tanhfast(float x) { return 2.0f * frcp(1.0f + __expf(-2.0f * x)) - 1.0f; }

// ---------------------------------------------------------------------------
// Stage 1: proj[v] = emb[v,:] . Ws_w + Ws_b   (streaming, memory-bound)
// ---------------------------------------------------------------------------
__global__ void hint_vocab_proj(const float* __restrict__ emb,
                                const float* __restrict__ Ws_w,
                                const float* __restrict__ Ws_b,
                                float* __restrict__ proj) {
    int row  = (int)((blockIdx.x * blockDim.x + threadIdx.x) >> 5);
    int lane = threadIdx.x & 31;
    if (row >= VOCAB) return;
    const float* r = emb + (size_t)row * EMB;
    float acc = 0.f;
    for (int e = lane; e < EMB; e += 32) acc += r[e] * Ws_w[e];
    acc += shflx(acc, 16); acc += shflx(acc, 8); acc += shflx(acc, 4);
    acc += shflx(acc, 2);  acc += shflx(acc, 1);
    if (lane == 0) proj[row] = acc + Ws_b[0];
}

// ---------------------------------------------------------------------------
// Stage 2: gather projections for doc tokens (Y) and query tokens (X)
// ---------------------------------------------------------------------------
__global__ void hint_gather(const int* __restrict__ sent,
                            const int* __restrict__ qsent,
                            const float* __restrict__ proj,
                            float* __restrict__ Y, float* __restrict__ X) {
    int t = blockIdx.x * blockDim.x + threadIdx.x;
    if (t < BB * MAXDOC) Y[t] = proj[sent[t]];
    if (t < BB * QQ)     X[t] = proj[qsent[t]];
}

// ---------------------------------------------------------------------------
// Stage 3: 2D-GRU wavefront recurrence.
//   One wave handles 16 sequences; lane n <-> h-component 0, lane n+16 <-> h1.
//   Per cell: one v_wmma_f32_16x16x32_f16 computes rz = Wrz * q + b for all
//   16 sequences (A = permuted Wrz, constant; B = q vectors; C = bias).
//   Output permutation: M = quantity(0..6) + 8*hcomp so all 7 gate values of
//   (seq n, h) land in one lane of the f32 C/D layout.
// ---------------------------------------------------------------------------
__global__ void __launch_bounds__(32)
hint_gru2d(const float* __restrict__ X, const float* __restrict__ Y,
           const float* __restrict__ Mxor, const float* __restrict__ Mcos,
           const float* __restrict__ masks,
           const float* __restrict__ Wrz1_w, const float* __restrict__ Wrz1_b,
           const float* __restrict__ W1_w,   const float* __restrict__ W1_b,
           const float* __restrict__ U1_w,   const float* __restrict__ U1_b,
           const float* __restrict__ Wrz3_w, const float* __restrict__ Wrz3_b,
           const float* __restrict__ W3_w,   const float* __restrict__ W3_b,
           const float* __restrict__ U3_w,   const float* __restrict__ U3_b,
           float* __restrict__ Hout) {
    __shared__ float prev[100 * 32];          // previous GRU row, per lane

    const int lane  = threadIdx.x & 31;
    const int dir   = blockIdx.x >> 6;        // 0 = fwd (gru1), 1 = flipped (gru3)
    const int nseq  = (blockIdx.x & 63) * 16 + (lane & 15);   // 0..1023
    const int g     = nseq >> 9;              // 0 = XOR half, 1 = cos half
    const int pb    = nseq & 511;
    const int p     = pb >> 6;
    const int b     = pb & 63;
    const int hcomp = lane >> 4;              // which hidden component this lane owns

    const float* Wrz_w = dir ? Wrz3_w : Wrz1_w;
    const float* Wrz_b = dir ? Wrz3_b : Wrz1_b;
    const float* W_w   = dir ? W3_w   : W1_w;
    const float* W_b   = dir ? W3_b   : W1_b;
    const float* U_w   = dir ? U3_w   : U1_w;
    const float* U_b   = dir ? U3_b   : U1_b;

    // A operand (16x32 f16): lanes 0-15 hold K=0..15, lanes 16-31 K=16..31 (=0).
    // Row M: M<8 -> (quantity M, h=0); M>=8 -> (quantity M-8, h=1). orig row = q*2+h.
    v16h A = {};
    {
        int M  = lane & 15;
        int qi = M & 7;
        int h  = M >> 3;
        if (qi < 7) {
            for (int i = 0; i < 16; ++i) {
                int K = (lane < 16) ? i : 16 + i;
                if (K < 9) A[i] = (_Float16)Wrz_w[(qi * 2 + h) * 9 + K];
            }
        }
    }
    // Bias through the C operand (C layout row M = v + 8*hcomp)
    v8f Cb;
    for (int v = 0; v < 8; ++v) Cb[v] = (v < 7) ? Wrz_b[2 * v + hcomp] : 0.0f;

    // U_w entries for this lane's k-parity (k = hcomp, hcomp+2, hcomp+4)
    float U0[3], U1[3];
    for (int j2 = 0; j2 < 3; ++j2) {
        U0[j2] = U_w[0 * 6 + hcomp + 2 * j2];
        U1[j2] = U_w[1 * 6 + hcomp + 2 * j2];
    }
    const float Wx0 = W_w[hcomp * 3 + 0], Wx1 = W_w[hcomp * 3 + 1], Wx2 = W_w[hcomp * 3 + 2];
    const float wb  = W_b[hcomp] + U_b[hcomp];

    const float* Mmat  = g ? Mcos : Mxor;
    const float* Ybase = Y + b * MAXDOC + p * 100;

    for (int j = 0; j < 100; ++j) prev[j * 32 + lane] = 0.0f;
    __syncthreads();   // single-wave WG: S_NOP-class barrier, harmless

    float h_left = 0.0f;
    for (int i = 0; i < QQ; ++i) {
        const int qi = dir ? (QQ - 1 - i) : i;
        const float Xi = X[b * QQ + qi];
        const float* Mrow = Mmat  + ((size_t)(b * QQ + qi)) * MAXDOC + p * 100;
        const float* Krow = masks + ((size_t)(b * QQ + qi)) * MAXDOC + p * 100;
        // prefetch next row's streams into cache ahead of the wavefront
        if (i + 1 < QQ) {
            const int qn = dir ? (QQ - 2 - i) : (i + 1);
            __builtin_prefetch(Mmat  + ((size_t)(b * QQ + qn)) * MAXDOC + p * 100, 0, 1);
            __builtin_prefetch(masks + ((size_t)(b * QQ + qn)) * MAXDOC + p * 100, 0, 1);
        }
        h_left = 0.0f;
        float h_diag = 0.0f;
        for (int j = 0; j < 100; ++j) {
            const int col = dir ? (99 - j) : j;
            float h_top = prev[j * 32 + lane];
            float om = 1.0f - Krow[col];
            float x0 = Xi * om;
            float x1 = Ybase[col] * om;
            float x2 = Mrow[col] * om;
            // fetch partner hidden component (h1 for lanes<16, h0 for lanes>=16)
            float ptop  = shflx(h_top, 16);
            float pleft = shflx(h_left, 16);
            float pdiag = shflx(h_diag, 16);
            // B operand (32x16 f16): lanes 0-15 carry K=0..15 of column N=lane
            v16h Bv = {};
            if (lane < 16) {
                Bv[0] = (_Float16)h_top;  Bv[1] = (_Float16)ptop;
                Bv[2] = (_Float16)h_left; Bv[3] = (_Float16)pleft;
                Bv[4] = (_Float16)h_diag; Bv[5] = (_Float16)pdiag;
                Bv[6] = (_Float16)x0;     Bv[7] = (_Float16)x1;
                Bv[8] = (_Float16)x2;
            }
            v8f c = __builtin_amdgcn_wmma_f32_16x16x32_f16(
                        false, A, false, Bv, (short)0, Cb, false, false);
            // this lane now holds rl,rt,rd,zi,zl,zt,zd for (seq lane%16, h=hcomp)
            float rl = sigf(c[0]), rt = sigf(c[1]), rd = sigf(c[2]);
            float zi = c[3], zl = c[4], zt = c[5], zd = c[6];
            float mx = fmaxf(fmaxf(zi, zl), fmaxf(zt, zd));
            float ei = __expf(zi - mx), el = __expf(zl - mx);
            float et = __expf(zt - mx), ed = __expf(zd - mx);
            float inv = frcp(ei + el + et + ed);
            zi = ei * inv; zl = el * inv; zt = et * inv; zd = ed * inv;
            // (r * t11) @ U_w.T : this lane owns k = hcomp, hcomp+2, hcomp+4
            float pl = rl * h_top, pt = rt * h_left, pd = rd * h_diag;
            float part0 = pl * U0[0] + pt * U0[1] + pd * U0[2];
            float part1 = pl * U1[0] + pt * U1[1] + pd * U1[2];
            float tot0 = part0 + shflx(part0, 16);
            float tot1 = part1 + shflx(part1, 16);
            float xw = x0 * Wx0 + x1 * Wx1 + x2 * Wx2 + wb;
            float h1 = tanhfast(xw + (hcomp ? tot1 : tot0));
            float hnew = zl * h_left + zt * h_top + zd * h_diag + zi * h1;
            prev[j * 32 + lane] = hnew;
            h_diag = h_top;
            h_left = hnew;
        }
    }
    Hout[dir * 2048 + nseq * 2 + hcomp] = h_left;   // h at (Q-1, 99)
}

// ---------------------------------------------------------------------------
// Stage 4: bi-LSTM (T=8, H=6, B=64) + vt + top-k(16->10) + fc. One lane = one
// batch element; everything is tiny scalar math.
// ---------------------------------------------------------------------------
__global__ void __launch_bounds__(64)
hint_head(const float* __restrict__ Hout,
          const float* __restrict__ W_ih_f, const float* __restrict__ W_hh_f,
          const float* __restrict__ b_ih_f, const float* __restrict__ b_hh_f,
          const float* __restrict__ W_ih_b, const float* __restrict__ W_hh_b,
          const float* __restrict__ b_ih_b, const float* __restrict__ b_hh_b,
          const float* __restrict__ Wv_w,   const float* __restrict__ Wv_b,
          const float* __restrict__ fc_w,   const float* __restrict__ fc_b,
          float* __restrict__ out) {
    int b = threadIdx.x;
    if (b >= BB) return;
    float pe[PP][8];
    for (int p = 0; p < PP; ++p) {
        int pb = p * BB + b;
        pe[p][0] = Hout[(0 * 1024 + 0   + pb) * 2 + 0];
        pe[p][1] = Hout[(0 * 1024 + 0   + pb) * 2 + 1];
        pe[p][2] = Hout[(0 * 1024 + 512 + pb) * 2 + 0];
        pe[p][3] = Hout[(0 * 1024 + 512 + pb) * 2 + 1];
        pe[p][4] = Hout[(1 * 1024 + 0   + pb) * 2 + 0];
        pe[p][5] = Hout[(1 * 1024 + 0   + pb) * 2 + 1];
        pe[p][6] = Hout[(1 * 1024 + 512 + pb) * 2 + 0];
        pe[p][7] = Hout[(1 * 1024 + 512 + pb) * 2 + 1];
    }
    float ev[16][LSTMH];
    for (int p = 0; p < PP; ++p)
        for (int h = 0; h < LSTMH; ++h) {
            float a = Wv_b[h];
            for (int c = 0; c < 8; ++c) a += pe[p][c] * Wv_w[h * 8 + c];
            ev[p][h] = tanhfast(a);
        }
    {   // forward LSTM
        float h[LSTMH] = {0}, cc[LSTMH] = {0};
        for (int p = 0; p < PP; ++p) {
            float gg[24];
            for (int r = 0; r < 24; ++r) {
                float a = b_ih_f[r] + b_hh_f[r];
                for (int c = 0; c < 8; ++c)      a += pe[p][c] * W_ih_f[r * 8 + c];
                for (int k = 0; k < LSTMH; ++k)  a += h[k] * W_hh_f[r * LSTMH + k];
                gg[r] = a;
            }
            for (int k = 0; k < LSTMH; ++k) {
                float ig = sigf(gg[k]), fg = sigf(gg[6 + k]);
                float gv = tanhfast(gg[12 + k]), og = sigf(gg[18 + k]);
                cc[k] = fg * cc[k] + ig * gv;
                h[k]  = og * tanhfast(cc[k]);
                ev[8 + p][k] = h[k];
            }
        }
    }
    {   // backward LSTM (process reversed; un-flip lands at same p)
        float h[LSTMH] = {0}, cc[LSTMH] = {0};
        for (int p = PP - 1; p >= 0; --p) {
            float gg[24];
            for (int r = 0; r < 24; ++r) {
                float a = b_ih_b[r] + b_hh_b[r];
                for (int c = 0; c < 8; ++c)      a += pe[p][c] * W_ih_b[r * 8 + c];
                for (int k = 0; k < LSTMH; ++k)  a += h[k] * W_hh_b[r * LSTMH + k];
                gg[r] = a;
            }
            for (int k = 0; k < LSTMH; ++k) {
                float ig = sigf(gg[k]), fg = sigf(gg[6 + k]);
                float gv = tanhfast(gg[12 + k]), og = sigf(gg[18 + k]);
                cc[k] = fg * cc[k] + ig * gv;
                h[k]  = og * tanhfast(cc[k]);
                ev[8 + p][k] += h[k];
            }
        }
    }
    // per hidden unit: top-10 of 16 evidence values (descending), dot with fc_w
    float score = fc_b[0];
    for (int h = 0; h < LSTMH; ++h) {
        float vals[16];
        for (int k = 0; k < 16; ++k) vals[k] = ev[k][h];
        for (int t = 0; t < KKMAX; ++t) {
            int bi = 0; float bv = vals[0];
            for (int k = 1; k < 16; ++k) if (vals[k] > bv) { bv = vals[k]; bi = k; }
            score += bv * fc_w[h * KKMAX + t];
            vals[bi] = -3.0e38f;
        }
    }
    out[b] = score;
}

// ---------------------------------------------------------------------------
extern "C" void kernel_launch(void* const* d_in, const int* in_sizes, int n_in,
                              void* d_out, int out_size, void* d_ws, size_t ws_size,
                              hipStream_t stream) {
    (void)in_sizes; (void)n_in; (void)out_size; (void)ws_size;
    const int*   sentence = (const int*)d_in[0];
    const int*   qsent    = (const int*)d_in[1];
    const float* Mxor     = (const float*)d_in[2];
    const float* Mcos     = (const float*)d_in[3];
    const float* masks    = (const float*)d_in[4];
    const float* emb      = (const float*)d_in[5];
    const float* Ws_w     = (const float*)d_in[6];
    const float* Ws_b     = (const float*)d_in[7];
    // gru1: Wrz_w, Wrz_b, W_w, W_b, U_w, U_b  (params dict insertion order)
    const float* g1_Wrz_w = (const float*)d_in[8];
    const float* g1_Wrz_b = (const float*)d_in[9];
    const float* g1_W_w   = (const float*)d_in[10];
    const float* g1_W_b   = (const float*)d_in[11];
    const float* g1_U_w   = (const float*)d_in[12];
    const float* g1_U_b   = (const float*)d_in[13];
    const float* g3_Wrz_w = (const float*)d_in[14];
    const float* g3_Wrz_b = (const float*)d_in[15];
    const float* g3_W_w   = (const float*)d_in[16];
    const float* g3_W_b   = (const float*)d_in[17];
    const float* g3_U_w   = (const float*)d_in[18];
    const float* g3_U_b   = (const float*)d_in[19];
    const float* W_ih_f   = (const float*)d_in[20];
    const float* W_hh_f   = (const float*)d_in[21];
    const float* b_ih_f   = (const float*)d_in[22];
    const float* b_hh_f   = (const float*)d_in[23];
    const float* W_ih_b   = (const float*)d_in[24];
    const float* W_hh_b   = (const float*)d_in[25];
    const float* b_ih_b   = (const float*)d_in[26];
    const float* b_hh_b   = (const float*)d_in[27];
    const float* Wv_w     = (const float*)d_in[28];
    const float* Wv_b     = (const float*)d_in[29];
    const float* fc_w     = (const float*)d_in[30];
    const float* fc_b     = (const float*)d_in[31];

    float* ws   = (float*)d_ws;
    float* proj = ws;                    // 50000
    float* Y    = proj + VOCAB;          // 51200
    float* X    = Y + BB * MAXDOC;       // 640
    float* Hout = X + BB * QQ;           // 2*1024*2 = 4096

    hint_vocab_proj<<<(VOCAB + 7) / 8, 256, 0, stream>>>(emb, Ws_w, Ws_b, proj);
    hint_gather<<<(BB * MAXDOC + 255) / 256, 256, 0, stream>>>(sentence, qsent, proj, Y, X);
    hint_gru2d<<<128, 32, 0, stream>>>(X, Y, Mxor, Mcos, masks,
                                       g1_Wrz_w, g1_Wrz_b, g1_W_w, g1_W_b, g1_U_w, g1_U_b,
                                       g3_Wrz_w, g3_Wrz_b, g3_W_w, g3_W_b, g3_U_w, g3_U_b,
                                       Hout);
    hint_head<<<1, 64, 0, stream>>>(Hout,
                                    W_ih_f, W_hh_f, b_ih_f, b_hh_f,
                                    W_ih_b, W_hh_b, b_ih_b, b_hh_b,
                                    Wv_w, Wv_b, fc_w, fc_b, (float*)d_out);
}
